// MLVAE_65163243815711
// MI455X (gfx1250) — matrix operand order repaired and checked by reading, not compiled, not run on hardware
//
#include <hip/hip_runtime.h>
#include <math.h>

// ---------------- problem constants ----------------
#define BB        8192
#define INDIM     2048
#define GG        256
#define VLATC     32
#define LOG2PI    1.8378770664093453f

// ---------------- WMMA vector types ----------------
typedef __attribute__((ext_vector_type(16))) __bf16 v16bf;
typedef __attribute__((ext_vector_type(8)))  __bf16 v8bf;
typedef __attribute__((ext_vector_type(8)))  float  v8f;

union FU { float f; unsigned u; };
union SB { unsigned short s; __bf16 b; };

__device__ __forceinline__ __bf16 f2bf(float f) {
  FU v; v.f = f;
  unsigned r = (v.u + 0x7FFFu + ((v.u >> 16) & 1u)) >> 16;  // RNE
  SB o; o.s = (unsigned short)r; return o.b;
}

// ---------------- tiny hash normal (stand-in for jax PRNG) ----------------
__device__ __forceinline__ unsigned pcg(unsigned v) {
  v = v * 747796405u + 2891336453u;
  unsigned w = ((v >> ((v >> 28) + 4u)) ^ v) * 277803737u;
  return (w >> 22) ^ w;
}
__device__ __forceinline__ float u01(unsigned h) {
  return (float)(h >> 8) * (1.0f / 16777216.0f) + (0.5f / 16777216.0f);
}
__device__ __forceinline__ float hashNormal(unsigned seed) {
  float u1 = u01(pcg(seed));
  float u2 = u01(pcg(seed ^ 0x9E3779B9u));
  return sqrtf(-2.0f * logf(u1)) * cosf(6.2831853071f * u2);
}

// ---------------- weight pack: fp32 KxN -> bf16 WMMA B-fragment layout ----
// layout: [kt][nt][lane][j] , lane in 0..31, j in 0..15
// lane&15 = column-in-tile, lane>>4 = K half, j = K offset within half
__global__ void pack_w(const float* __restrict__ W, __bf16* __restrict__ out,
                       int K, int N) {
  int idx = blockIdx.x * blockDim.x + threadIdx.x;
  int total = K * N;
  if (idx >= total) return;
  int t    = idx >> 9;          // tile index (512 elems/tile)
  int r    = idx & 511;
  int lane = r >> 4;
  int j    = r & 15;
  int ntiles = N >> 4;
  int kt = t / ntiles, nt = t % ntiles;
  int col = nt * 16 + (lane & 15);
  int kk  = kt * 32 + (lane >> 4) * 16 + j;
  out[idx] = f2bf(W[(size_t)kk * N + col]);
}

// ---------------- fp32 -> bf16 activation convert ----------------
__global__ void f32_to_bf16(const float* __restrict__ in, __bf16* __restrict__ out, int n) {
  int i = blockIdx.x * blockDim.x + threadIdx.x;
  if (i < n) out[i] = f2bf(in[i]);
}

// ---------------- zero fill ----------------
__global__ void fillz(float* p, int n) {
  int i = blockIdx.x * blockDim.x + threadIdx.x;
  if (i < n) p[i] = 0.0f;
}

// ---------------- WMMA GEMM: C = A(MxK,bf16) @ Wpacked + bias ------------
// block = 256 threads = 8 waves. Each wave computes a 16x64 strip (4 WMMA
// accumulators), so one A-fragment load is shared by 4 WMMAs (A reuse x4).
// All layer widths are multiples of 64 -> a strip never straddles N.
__global__ void wmma_gemm(const __bf16* __restrict__ A,
                          const __bf16* __restrict__ Bp,
                          const float*  __restrict__ bias,
                          float* __restrict__ Cf, __bf16* __restrict__ Cb,
                          int M, int N, int K) {
  const int lane   = threadIdx.x & 31;
  const int wave   = threadIdx.x >> 5;
  const int strip  = blockIdx.x * 8 + wave;   // 64-wide N strip
  const int ntile0 = strip * 4;
  if (ntile0 * 16 >= N) return;
  const int mtile = blockIdx.y;

  const int h    = lane >> 4;                 // wave half
  const int mrow = mtile * 16 + (lane & 15);
  const int ncol = lane & 15;
  const int ntiles = N >> 4;

  v8f acc[4];
#pragma unroll
  for (int t = 0; t < 4; ++t) {
    float bv = bias[(ntile0 + t) * 16 + ncol];
#pragma unroll
    for (int i = 0; i < 8; ++i) acc[t][i] = bv;
  }

  const __bf16* arow = A + (size_t)mrow * K;
  const size_t  kstride = (size_t)ntiles * 512;      // packed elems per k-tile
  const __bf16* bbase = Bp + (size_t)ntile0 * 512 + lane * 16;

  for (int k0 = 0; k0 < K; k0 += 32) {
    // A fragment (16x32 bf16): lanes 0-15 hold K {0..15}, lanes 16-31 {16..31}
    v8bf alo = *(const v8bf*)(arow + k0 + h * 16);
    v8bf ahi = *(const v8bf*)(arow + k0 + h * 16 + 8);
    v16bf af;
#pragma unroll
    for (int i = 0; i < 8; ++i) { af[i] = alo[i]; af[8 + i] = ahi[i]; }

    const __bf16* bpk = bbase + (size_t)(k0 >> 5) * kstride;
    // prefetch B two k-tiles ahead (speculative; lowers to global_prefetch_b8)
    __builtin_prefetch(bpk + 2 * kstride, 0, 1);
#pragma unroll
    for (int t = 0; t < 4; ++t) {
      v16bf bfm = *(const v16bf*)(bpk + t * 512);
      acc[t] = __builtin_amdgcn_wmma_f32_16x16x32_bf16(false, af, false, bfm,
                                                       (short)0, acc[t], false, false);
    }
  }

#pragma unroll
  for (int t = 0; t < 4; ++t) {
    const int col = (ntile0 + t) * 16 + ncol;
#pragma unroll
    for (int r = 0; r < 8; ++r) {
      int orow = mtile * 16 + r + 8 * h;   // C layout: VGPR r -> M=r (+8 upper half)
      if (Cf) Cf[(size_t)orow * N + col] = acc[t][r];
      if (Cb) Cb[(size_t)orow * N + col] = f2bf(acc[t][r]);
    }
  }
}

// ---------------- LayerNorm + ReLU, fp32 in -> bf16 out ----------------
__global__ void ln_relu(const float* __restrict__ in, const float* __restrict__ gamma,
                        const float* __restrict__ beta, __bf16* __restrict__ out, int N) {
  __shared__ float red[256];
  const int row = blockIdx.x;
  const float* p = in + (size_t)row * N;

  float s = 0.f;
  for (int c = threadIdx.x; c < N; c += 256) s += p[c];
  red[threadIdx.x] = s; __syncthreads();
  for (int off = 128; off > 0; off >>= 1) {
    if (threadIdx.x < off) red[threadIdx.x] += red[threadIdx.x + off];
    __syncthreads();
  }
  float mean = red[0] / (float)N;
  __syncthreads();

  float q = 0.f;
  for (int c = threadIdx.x; c < N; c += 256) { float d = p[c] - mean; q += d * d; }
  red[threadIdx.x] = q; __syncthreads();
  for (int off = 128; off > 0; off >>= 1) {
    if (threadIdx.x < off) red[threadIdx.x] += red[threadIdx.x + off];
    __syncthreads();
  }
  float rstd = rsqrtf(red[0] / (float)N + 1e-5f);

  __bf16* o = out + (size_t)row * N;
  for (int c = threadIdx.x; c < N; c += 256) {
    float y = (p[c] - mean) * rstd * gamma[c] + beta[c];
    o[c] = f2bf(fmaxf(y, 0.f));
  }
}

// ---------------- head: copy mu/logvar to out, segment-sum precision ------
__global__ void head_post(const float* __restrict__ v, const float* __restrict__ e,
                          const int* __restrict__ labels,
                          float* __restrict__ o_vm, float* __restrict__ o_vlv,
                          float* __restrict__ o_em, float* __restrict__ o_elv,
                          float* __restrict__ sum_prec, float* __restrict__ sum_wmu) {
  int idx = blockIdx.x * blockDim.x + threadIdx.x;
  if (idx >= BB * VLATC) return;
  int i = idx >> 5, j = idx & 31;
  float vm  = v[(size_t)i * 64 + j];
  float vlv = v[(size_t)i * 64 + 32 + j];
  float em  = e[(size_t)i * 64 + j];
  float elv = e[(size_t)i * 64 + 32 + j];
  o_vm[idx] = vm; o_vlv[idx] = vlv; o_em[idx] = em; o_elv[idx] = elv;
  float var = expf(vlv);
  if (var == 0.f) var = 1e-6f;
  float prec = 1.f / var;
  int lab = labels[i];
  atomicAdd(&sum_prec[lab * VLATC + j], prec);
  atomicAdd(&sum_wmu[lab * VLATC + j], vm * prec);
}

// ---------------- group finalize + group eps ----------------
__global__ void group_fin(const float* __restrict__ sum_prec,
                          const float* __restrict__ sum_wmu,
                          float* __restrict__ gmu, float* __restrict__ gvar,
                          float* __restrict__ eps) {
  int idx = blockIdx.x * blockDim.x + threadIdx.x;
  if (idx >= GG * VLATC) return;
  float gv = 1.f / sum_prec[idx];
  gmu[idx]  = sum_wmu[idx] * gv;
  gvar[idx] = (gv == 0.f) ? 1e-6f : gv;
  eps[idx]  = hashNormal(0xABCD1234u + (unsigned)idx);
}

// ---------------- build latent z (bf16) ----------------
__global__ void build_z(const int* __restrict__ labels,
                        const float* __restrict__ gmu, const float* __restrict__ gvar,
                        const float* __restrict__ eps,
                        const float* __restrict__ o_em, const float* __restrict__ o_elv,
                        __bf16* __restrict__ z) {
  int idx = blockIdx.x * blockDim.x + threadIdx.x;
  if (idx >= BB * 64) return;
  int i = idx >> 6, c = idx & 63;
  float zz;
  if (c < 32) {
    int lab = labels[i];
    float gv = gvar[lab * VLATC + c];
    zz = eps[lab * VLATC + c] * sqrtf(gv) + gmu[lab * VLATC + c];
  } else {
    int j = c - 32;
    float elv = o_elv[i * VLATC + j];
    float em  = o_em[i * VLATC + j];
    zz = hashNormal(0x51A7E000u + (unsigned)(i * 32 + j)) * expf(0.5f * elv) + em;
  }
  z[idx] = f2bf(zz);
}

// ---------------- reconstruction loss ----------------
__global__ void loss_accum(const float* __restrict__ x, const float* __restrict__ rec,
                           float* __restrict__ sum, int n) {
  __shared__ float red[256];
  float s = 0.f;
  for (int i = blockIdx.x * blockDim.x + threadIdx.x; i < n; i += gridDim.x * blockDim.x) {
    float d = x[i] - rec[i];
    s += d * d;
  }
  red[threadIdx.x] = s; __syncthreads();
  for (int off = 128; off > 0; off >>= 1) {
    if (threadIdx.x < off) red[threadIdx.x] += red[threadIdx.x + off];
    __syncthreads();
  }
  if (threadIdx.x == 0) atomicAdd(sum, red[0]);
}

__global__ void loss_final(const float* __restrict__ sum, float* __restrict__ out) {
  if (threadIdx.x == 0 && blockIdx.x == 0)
    out[0] = -0.5f * (sum[0] / (float)((size_t)BB * INDIM) + LOG2PI);
}

// =============================================================================
extern "C" void kernel_launch(void* const* d_in, const int* in_sizes, int n_in,
                              void* d_out, int out_size, void* d_ws, size_t ws_size,
                              hipStream_t stream) {
  (void)in_sizes; (void)n_in; (void)out_size; (void)ws_size;

  // ---- inputs (setup_inputs dict order, params nested in insertion order) ----
  const float* x      = (const float*)d_in[0];
  const int*   labels = (const int*)d_in[1];
  // weights: index, K, N   biases/gamma/beta follow each W
  const int wi[12] = { 2, 6, 10, 14, 18, 20, 22, 24, 28, 32, 36, 40 };
  const int wK[12] = { 2048, 2048, 1024, 512, 256, 128, 128,  64, 256,  512, 1024, 2048 };
  const int wN[12] = { 2048, 1024,  512, 256, 128,  64,  64, 256, 512, 1024, 2048, 2048 };

  // ---- workspace carve ----
  char* base = (char*)d_ws;
  size_t off = 0;
  auto carve = [&](size_t bytes) -> char* {
    char* p = base + off;
    off += (bytes + 255) & ~(size_t)255;
    return p;
  };
  __bf16* wp[12];
  for (int i = 0; i < 12; ++i)
    wp[i] = (__bf16*)carve((size_t)wK[i] * wN[i] * sizeof(__bf16));
  float*  actF  = (float*) carve((size_t)BB * 2048 * sizeof(float));
  __bf16* bufA  = (__bf16*)carve((size_t)BB * 2048 * sizeof(__bf16));
  __bf16* bufB  = (__bf16*)carve((size_t)BB * 2048 * sizeof(__bf16));
  float*  vbuf  = (float*) carve((size_t)BB * 64 * sizeof(float));
  float*  ebuf  = (float*) carve((size_t)BB * 64 * sizeof(float));
  float*  sprec = (float*) carve((size_t)GG * VLATC * sizeof(float));
  float*  swmu  = (float*) carve((size_t)GG * VLATC * sizeof(float));
  float*  gmu   = (float*) carve((size_t)GG * VLATC * sizeof(float));
  float*  gvar  = (float*) carve((size_t)GG * VLATC * sizeof(float));
  float*  epsg  = (float*) carve((size_t)GG * VLATC * sizeof(float));
  __bf16* zbuf  = (__bf16*)carve((size_t)BB * 64 * sizeof(__bf16));
  float*  lsum  = (float*) carve(256);

  // ---- output carve (return order: vm, vlv, em, elv, rec_term, rec_x) ----
  float* out     = (float*)d_out;
  float* o_vm    = out;
  float* o_vlv   = out + (size_t)BB * VLATC;
  float* o_em    = out + (size_t)BB * VLATC * 2;
  float* o_elv   = out + (size_t)BB * VLATC * 3;
  float* o_rterm = out + (size_t)BB * VLATC * 4;
  float* o_recx  = o_rterm + 1;

  // ---- 1) pack weights to bf16 WMMA layout ----
  for (int i = 0; i < 12; ++i) {
    int total = wK[i] * wN[i];
    pack_w<<<(total + 255) / 256, 256, 0, stream>>>(
        (const float*)d_in[wi[i]], wp[i], wK[i], wN[i]);
  }

  // ---- 2) x -> bf16 ----
  {
    int n = BB * INDIM;
    f32_to_bf16<<<(n + 255) / 256, 256, 0, stream>>>(x, bufA, n);
  }

  auto gemm = [&](const __bf16* A, int widx, const float* bias,
                  float* Cf, __bf16* Cb, int M, int N, int K) {
    int strips = N / 64;                       // 16x64 strip per wave
    dim3 grid((strips + 7) / 8, M / 16);
    wmma_gemm<<<grid, 256, 0, stream>>>(A, wp[widx], bias, Cf, Cb, M, N, K);
  };

  // ---- 3) encoder MLP: 4x (GEMM+bias -> LN -> ReLU) ----
  // layer i: W at wi[i], b at wi[i]+1, gamma wi[i]+2, beta wi[i]+3
  __bf16* cur = bufA;
  __bf16* nxt = bufB;
  const int encK[4] = { 2048, 2048, 1024, 512 };
  const int encN[4] = { 2048, 1024,  512, 256 };
  for (int l = 0; l < 4; ++l) {
    gemm(cur, l, (const float*)d_in[wi[l] + 1], actF, nullptr, BB, encN[l], encK[l]);
    ln_relu<<<BB, 256, 0, stream>>>(actF, (const float*)d_in[wi[l] + 2],
                                    (const float*)d_in[wi[l] + 3], nxt, encN[l]);
    __bf16* t = cur; cur = nxt; nxt = t;
  }

  // ---- 4) enc_lin (256->128): bf16-only output feeds two head GEMMs ----
  gemm(cur, 4, (const float*)d_in[19], nullptr, nxt, BB, 128, 256);

  // ---- 5) heads ----
  gemm(nxt, 5, (const float*)d_in[21], vbuf, nullptr, BB, 64, 128);  // variete
  gemm(nxt, 6, (const float*)d_in[23], ebuf, nullptr, BB, 64, 128);  // env

  // ---- 6) group evidence + sampling ----
  fillz<<<(2 * GG * VLATC + 255) / 256, 256, 0, stream>>>(sprec, 2 * GG * VLATC);
  {
    int n = BB * VLATC;
    head_post<<<(n + 255) / 256, 256, 0, stream>>>(vbuf, ebuf, labels,
                                                   o_vm, o_vlv, o_em, o_elv,
                                                   sprec, swmu);
  }
  group_fin<<<(GG * VLATC + 255) / 256, 256, 0, stream>>>(sprec, swmu, gmu, gvar, epsg);
  {
    int n = BB * 64;
    build_z<<<(n + 255) / 256, 256, 0, stream>>>(labels, gmu, gvar, epsg,
                                                 o_em, o_elv, zbuf);
  }

  // ---- 7) decoder MLP ----
  const int decK[4] = { 64, 256, 512, 1024 };
  const int decN[4] = { 256, 512, 1024, 2048 };
  cur = zbuf; nxt = bufA;
  for (int l = 0; l < 4; ++l) {
    int widx = 7 + l;
    gemm(cur, widx, (const float*)d_in[wi[widx] + 1], actF, nullptr, BB, decN[l], decK[l]);
    ln_relu<<<BB, 256, 0, stream>>>(actF, (const float*)d_in[wi[widx] + 2],
                                    (const float*)d_in[wi[widx] + 3], nxt, decN[l]);
    cur = nxt;
    nxt = (cur == bufA) ? bufB : bufA;
  }

  // ---- 8) dec_lin -> rec_x (fp32 straight into d_out) ----
  gemm(cur, 11, (const float*)d_in[41], o_recx, nullptr, BB, INDIM, 2048);

  // ---- 9) reconstruction term ----
  fillz<<<1, 256, 0, stream>>>(lsum, 1);
  loss_accum<<<4096, 256, 0, stream>>>(x, o_recx, lsum, BB * INDIM);
  loss_final<<<1, 1, 0, stream>>>(lsum, o_rterm);
}
